// ConvMambaBlock_11836929868493
// MI455X (gfx1250) — compile-verified, matrix operand order checked by reading, not run on hardware
//
#include <hip/hip_runtime.h>
#include <cstdint>
#include <cstddef>

// ---------------- problem constants ----------------
#define BDIM   8
#define LDIM   2048
#define DDIM   512
#define TTOK   (BDIM * LDIM)   // 16384 tokens
#define DI2    2048            // 2*DI (mamba in_proj out, ccm w1 out)
#define DIC    1024            // DI == INNER
#define NSTATE 16
#define DTRNK  32
#define KCONV  31

typedef __attribute__((ext_vector_type(16))) __bf16 v16bf;
typedef __attribute__((ext_vector_type(8)))  float  v8f;

struct __attribute__((aligned(16))) U4 { unsigned int a, b, c, d; };
struct __attribute__((aligned(16))) F4 { float x, y, z, w; };

__device__ __forceinline__ unsigned short f2bf(float f) {
  unsigned int u = __float_as_uint(f);
  unsigned int r = u + 0x7FFFu + ((u >> 16) & 1u);  // round-to-nearest-even
  return (unsigned short)(r >> 16);
}
__device__ __forceinline__ float silu_f(float v) { return v / (1.f + __expf(-v)); }
__device__ __forceinline__ float act_apply(float v, int act) {
  if (act == 1) return silu_f(v);
  if (act == 2) return (v > 20.f) ? v : log1pf(__expf(v));  // softplus
  return v;
}

// ---- gfx1250 async global->LDS copy (per-lane 16B), tracked on ASYNCcnt ----
__device__ __forceinline__ void async_copy_b128(unsigned lds_off, const void* gaddr) {
  asm volatile("global_load_async_to_lds_b128 %0, %1, off"
               :: "v"(lds_off), "v"(gaddr) : "memory");
}
__device__ __forceinline__ void wait_async0() {
  asm volatile("s_wait_asynccnt 0x0" ::: "memory");
}

// ---------------- fp32 -> bf16 weight mirror ----------------
__global__ void cvt_bf16_kernel(const float* __restrict__ in,
                                unsigned short* __restrict__ out, int n) {
  int i = blockIdx.x * blockDim.x + threadIdx.x;
  if (i < n) out[i] = f2bf(in[i]);
}

// ---------------- LayerNorm: one wave32 per token (D = 512) ----------------
__global__ __launch_bounds__(256) void ln_kernel(const float* __restrict__ x,
                                                 const float* __restrict__ g,
                                                 const float* __restrict__ b,
                                                 float* __restrict__ y) {
  int wave = threadIdx.x >> 5, lane = threadIdx.x & 31;
  int t = blockIdx.x * 8 + wave;
  if (t >= TTOK) return;
  const float* xr = x + (size_t)t * DDIM;
  float vals[16];
  float s = 0.f;
#pragma unroll
  for (int i = 0; i < 16; ++i) { vals[i] = xr[lane + 32 * i]; s += vals[i]; }
#pragma unroll
  for (int off = 16; off > 0; off >>= 1) s += __shfl_xor(s, off, 32);
  float mean = s * (1.f / DDIM);
  float vs = 0.f;
#pragma unroll
  for (int i = 0; i < 16; ++i) { float d = vals[i] - mean; vs += d * d; }
#pragma unroll
  for (int off = 16; off > 0; off >>= 1) vs += __shfl_xor(vs, off, 32);
  float inv = rsqrtf(vs * (1.f / DDIM) + 1e-5f);
  float* yr = y + (size_t)t * DDIM;
#pragma unroll
  for (int i = 0; i < 16; ++i) {
    int c = lane + 32 * i;
    yr[c] = (vals[i] - mean) * inv * g[c] + b[c];
  }
}

// ---------------- WMMA bf16 GEMM ----------------
// C[m][n] = src[m][n] + alpha * act( sum_k A[m][k]*W[n][k] + bias[n] )
// A: fp32 [M][lda]; W: bf16 ushort [N][K] row-major; tile 128x128x32, 8 waves.
// B tile staged via global_load_async_to_lds_b128 (ASYNCcnt); A converted via VGPRs.
__global__ __launch_bounds__(256) void gemm_bf16_wmma(
    const float* __restrict__ A, int lda,
    const unsigned short* __restrict__ Bw,
    const float* __restrict__ bias,   // nullable
    const float* __restrict__ src,    // nullable residual source [M][ldc]
    float* __restrict__ C, int ldc,
    int M, int N, int K, float alpha, int act) {
  __shared__ unsigned short As[128 * 40];  // padded stride: 40 ushorts (80B)
  __shared__ unsigned short Bs[128 * 40];

  const int bm = blockIdx.y * 128, bn = blockIdx.x * 128;
  const int tid = threadIdx.x;
  const int wave = tid >> 5, lane = tid & 31;
  const int wm = wave >> 2, wn = wave & 3;  // 2 x 4 wave grid
  const int rl = lane & 15, kbase = (lane >> 4) * 8;

  v8f acc[4][2];
#pragma unroll
  for (int i = 0; i < 4; ++i)
#pragma unroll
    for (int j = 0; j < 2; ++j) acc[i][j] = (v8f)(0.f);

  for (int k0 = 0; k0 < K; k0 += 32) {
    // stage B first: async global->LDS (overlaps with A conversion below)
#pragma unroll
    for (int i = 0; i < 2; ++i) {
      int c = tid + 256 * i;                // 0..511 uint4 chunks
      int r = c >> 2, c8 = (c & 3) * 8;
      int n = bn + r;
      unsigned short* lp = &Bs[r * 40 + c8];
      if (n < N) {
        async_copy_b128((unsigned)(size_t)lp, Bw + (size_t)n * K + k0 + c8);
      } else {
        U4 z = {0u, 0u, 0u, 0u};
        *(U4*)lp = z;                       // zero-fill past N (ds_store_b128)
      }
    }
    // stage A (128x32 fp32 -> bf16 in LDS) through VGPRs
#pragma unroll
    for (int i = 0; i < 4; ++i) {
      int c = tid + 256 * i;                // 0..1023 float4 chunks
      int r = c >> 3, c4 = (c & 7) * 4;
      F4 f = *(const F4*)(A + (size_t)(bm + r) * lda + k0 + c4);
      unsigned short* dst = &As[r * 40 + c4];
      dst[0] = f2bf(f.x); dst[1] = f2bf(f.y);
      dst[2] = f2bf(f.z); dst[3] = f2bf(f.w);
    }
    // prefetch next K tile of A (global_prefetch_b8)
    if (k0 + 32 < K) {
      const float* pf = A + (size_t)(bm + (tid >> 1)) * lda + (k0 + 32) + (tid & 1) * 16;
      __builtin_prefetch(pf, 0, 1);
    }
    wait_async0();       // this wave's async B chunks landed in LDS
    __syncthreads();     // all waves' chunks visible

    v16bf afr[4], bfr[2];
    union Frag { U4 q[2]; v16bf v; };
#pragma unroll
    for (int fm = 0; fm < 4; ++fm) {
      int r = wm * 64 + fm * 16 + rl;
      Frag fa;
      fa.q[0] = *(const U4*)&As[r * 40 + kbase];
      fa.q[1] = *(const U4*)&As[r * 40 + kbase + 16];
      afr[fm] = fa.v;
    }
#pragma unroll
    for (int fn = 0; fn < 2; ++fn) {
      int r = wn * 32 + fn * 16 + rl;
      Frag fb;
      fb.q[0] = *(const U4*)&Bs[r * 40 + kbase];
      fb.q[1] = *(const U4*)&Bs[r * 40 + kbase + 16];
      bfr[fn] = fb.v;
    }
#pragma unroll
    for (int fm = 0; fm < 4; ++fm)
#pragma unroll
      for (int fn = 0; fn < 2; ++fn)
        acc[fm][fn] = __builtin_amdgcn_wmma_f32_16x16x32_bf16(
            false, afr[fm], false, bfr[fn], (short)0, acc[fm][fn], false, false);
    __syncthreads();
  }

  // epilogue
  const int hi = lane >> 4;
#pragma unroll
  for (int fm = 0; fm < 4; ++fm) {
#pragma unroll
    for (int fn = 0; fn < 2; ++fn) {
      int col = bn + wn * 32 + fn * 16 + rl;
      if (col >= N) continue;
      float bv = bias ? bias[col] : 0.f;
#pragma unroll
      for (int r = 0; r < 8; ++r) {
        int row = bm + wm * 64 + fm * 16 + hi * 8 + r;
        float v = acc[fm][fn][r] + bv;
        v = act_apply(v, act) * alpha;
        float s = src ? src[(size_t)row * ldc + col] : 0.f;
        C[(size_t)row * ldc + col] = s + v;
      }
    }
  }
}

// ---------------- Mamba depthwise conv (DCONV=4) + SiLU ----------------
// dir=0: causal (left pad 3); dir=1: equivalent of flip-conv-flip (right pad 3)
__global__ void mamba_conv_silu(const float* __restrict__ xz,   // [T, 2048], u = cols 0..1023
                                const float* __restrict__ w,    // [1024][4]
                                const float* __restrict__ bias, // [1024]
                                float* __restrict__ out,        // [T, 1024]
                                int dir) {
  int idx = blockIdx.x * blockDim.x + threadIdx.x;
  if (idx >= TTOK * DIC) return;
  int d = idx & (DIC - 1);
  int t = (idx >> 10) & (LDIM - 1);
  int b = idx >> 21;
  float acc = bias[d];
#pragma unroll
  for (int j = 0; j < 4; ++j) {
    int tt = dir ? (t + 3 - j) : (t - 3 + j);
    if (tt >= 0 && tt < LDIM)
      acc += w[d * 4 + j] * xz[((size_t)b * LDIM + tt) * DI2 + d];
  }
  out[((size_t)b * LDIM + t) * DIC + d] = silu_f(acc);
}

// ---------------- selective scan: one lane per (b, channel) ----------------
__global__ void mamba_scan_kernel(const float* __restrict__ dt,    // [T,1024]
                                  const float* __restrict__ xdb,   // [T,64]: dtR|B|C
                                  const float* __restrict__ u,     // [T,1024]
                                  const float* __restrict__ xz,    // [T,2048], z = cols 1024..
                                  const float* __restrict__ A_log, // [1024][16]
                                  const float* __restrict__ Dp,    // [1024]
                                  float* __restrict__ y,           // [T,1024]
                                  int dir) {
  int idx = blockIdx.x * blockDim.x + threadIdx.x;
  if (idx >= BDIM * DIC) return;
  int d = idx & (DIC - 1);
  int b = idx >> 10;
  float An[NSTATE], h[NSTATE];
#pragma unroll
  for (int n = 0; n < NSTATE; ++n) { An[n] = -__expf(A_log[d * NSTATE + n]); h[n] = 0.f; }
  float dparam = Dp[d];
  for (int s = 0; s < LDIM; ++s) {
    int t = dir ? (LDIM - 1 - s) : s;
    size_t row = (size_t)b * LDIM + t;
    float dtv = dt[row * DIC + d];
    float uv = u[row * DIC + d];
    const float* Bv = xdb + row * 64 + DTRNK;
    const float* Cv = Bv + NSTATE;
    float du = dtv * uv;
    float acc = 0.f;
#pragma unroll
    for (int n = 0; n < NSTATE; ++n) {
      h[n] = __expf(dtv * An[n]) * h[n] + du * Bv[n];
      acc += h[n] * Cv[n];
    }
    float zv = xz[row * DI2 + DIC + d];
    y[row * DIC + d] = (acc + uv * dparam) * silu_f(zv);
  }
}

// ---------------- CCM GLU: a * sigmoid(g) ----------------
__global__ void glu_kernel(const float* __restrict__ hh,  // [T, 2048]
                           float* __restrict__ out) {     // [T, 1024]
  int idx = blockIdx.x * blockDim.x + threadIdx.x;
  if (idx >= TTOK * DIC) return;
  int d = idx & (DIC - 1);
  size_t row = (size_t)(idx >> 10);
  float a = hh[row * DI2 + d];
  float g = hh[row * DI2 + DIC + d];
  out[row * DIC + d] = a / (1.f + __expf(-g));  // a * sigmoid(g)
}

// ---------------- CCM depthwise conv K=31 (same pad) + BN + SiLU ----------------
__global__ void ccm_conv_bn_silu(const float* __restrict__ in,   // [T,1024]
                                 const float* __restrict__ w,    // [1024][31]
                                 const float* __restrict__ wb,   // [1024]
                                 const float* __restrict__ bng, const float* __restrict__ bnb,
                                 const float* __restrict__ bnm, const float* __restrict__ bnv,
                                 float* __restrict__ out) {
  int idx = blockIdx.x * blockDim.x + threadIdx.x;
  if (idx >= TTOK * DIC) return;
  int d = idx & (DIC - 1);
  int t = (idx >> 10) & (LDIM - 1);
  int b = idx >> 21;
  float acc = wb[d];
#pragma unroll 8
  for (int j = 0; j < KCONV; ++j) {
    int tt = t - (KCONV - 1) / 2 + j;
    if (tt >= 0 && tt < LDIM)
      acc += w[d * KCONV + j] * in[((size_t)b * LDIM + tt) * DIC + d];
  }
  float v = (acc - bnm[d]) * rsqrtf(bnv[d] + 1e-5f) * bng[d] + bnb[d];
  out[((size_t)b * LDIM + t) * DIC + d] = silu_f(v);
}

// ================= host orchestration =================
static inline size_t al256(size_t x) { return (x + 255) & ~(size_t)255; }

extern "C" void kernel_launch(void* const* d_in, const int* in_sizes, int n_in,
                              void* d_out, int out_size, void* d_ws, size_t ws_size,
                              hipStream_t stream) {
  (void)in_sizes; (void)n_in; (void)out_size; (void)ws_size;
  // -------- input pointer map (setup_inputs flatten order) --------
  const float* x = (const float*)d_in[0];
  const float* ffm_g[2]  = {(const float*)d_in[1],  (const float*)d_in[7]};
  const float* ffm_b[2]  = {(const float*)d_in[2],  (const float*)d_in[8]};
  const float* ffm_w1[2] = {(const float*)d_in[3],  (const float*)d_in[9]};
  const float* ffm_b1[2] = {(const float*)d_in[4],  (const float*)d_in[10]};
  const float* ffm_w2[2] = {(const float*)d_in[5],  (const float*)d_in[11]};
  const float* ffm_b2[2] = {(const float*)d_in[6],  (const float*)d_in[12]};
  const float* m_ln_g = (const float*)d_in[13];
  const float* m_ln_b = (const float*)d_in[14];
  const float* mm_in_w[2]   = {(const float*)d_in[15], (const float*)d_in[24]};
  const float* mm_conv_w[2] = {(const float*)d_in[16], (const float*)d_in[25]};
  const float* mm_conv_b[2] = {(const float*)d_in[17], (const float*)d_in[26]};
  const float* mm_xproj[2]  = {(const float*)d_in[18], (const float*)d_in[27]};
  const float* mm_dt_w[2]   = {(const float*)d_in[19], (const float*)d_in[28]};
  const float* mm_dt_b[2]   = {(const float*)d_in[20], (const float*)d_in[29]};
  const float* mm_Alog[2]   = {(const float*)d_in[21], (const float*)d_in[30]};
  const float* mm_D[2]      = {(const float*)d_in[22], (const float*)d_in[31]};
  const float* mm_out_w[2]  = {(const float*)d_in[23], (const float*)d_in[32]};
  const float* c_g  = (const float*)d_in[33]; const float* c_b  = (const float*)d_in[34];
  const float* c_w1 = (const float*)d_in[35]; const float* c_b1 = (const float*)d_in[36];
  const float* c_dw = (const float*)d_in[37]; const float* c_dwb= (const float*)d_in[38];
  const float* c_bng= (const float*)d_in[39]; const float* c_bnb= (const float*)d_in[40];
  const float* c_bnm= (const float*)d_in[41]; const float* c_bnv= (const float*)d_in[42];
  const float* c_w2 = (const float*)d_in[43]; const float* c_b2 = (const float*)d_in[44];
  const float* post_g = (const float*)d_in[45];
  const float* post_b = (const float*)d_in[46];
  float* out = (float*)d_out;

  // -------- workspace carve --------
  uint8_t* wsp = (uint8_t*)d_ws;
  size_t off = 0;
  auto alloc = [&](size_t bytes) -> void* { void* p = wsp + off; off = al256(off + bytes); return p; };
  float* resid   = (float*)alloc((size_t)TTOK * DDIM * 4);
  float* lnbuf   = (float*)alloc((size_t)TTOK * DDIM * 4);
  float* buf2048 = (float*)alloc((size_t)TTOK * DI2 * 4);
  float* b1024a  = (float*)alloc((size_t)TTOK * DIC * 4);
  float* b1024b  = (float*)alloc((size_t)TTOK * DIC * 4);
  float* b1024c  = (float*)alloc((size_t)TTOK * DIC * 4);
  float* bufS    = (float*)alloc((size_t)TTOK * 64 * 4);
  auto allocW = [&](size_t n) -> unsigned short* { return (unsigned short*)alloc(n * 2); };
  unsigned short* w_ffm_w1[2], *w_ffm_w2[2], *w_in_w[2], *w_xproj[2], *w_dt_w[2], *w_out_w[2];
  for (int i = 0; i < 2; ++i) {
    w_ffm_w1[i] = allocW((size_t)DI2 * DDIM);
    w_ffm_w2[i] = allocW((size_t)DDIM * DI2);
    w_in_w[i]   = allocW((size_t)DI2 * DDIM);
    w_xproj[i]  = allocW((size_t)64 * DIC);
    w_dt_w[i]   = allocW((size_t)DIC * DTRNK);
    w_out_w[i]  = allocW((size_t)DDIM * DIC);
  }
  unsigned short* w_c_w1 = allocW((size_t)DI2 * DDIM);
  unsigned short* w_c_w2 = allocW((size_t)DDIM * DIC);

  auto cvt = [&](const float* src, unsigned short* dst, size_t n) {
    cvt_bf16_kernel<<<(unsigned)((n + 255) / 256), 256, 0, stream>>>(src, dst, (int)n);
  };
  for (int i = 0; i < 2; ++i) {
    cvt(ffm_w1[i], w_ffm_w1[i], (size_t)DI2 * DDIM);
    cvt(ffm_w2[i], w_ffm_w2[i], (size_t)DDIM * DI2);
    cvt(mm_in_w[i], w_in_w[i], (size_t)DI2 * DDIM);
    cvt(mm_xproj[i], w_xproj[i], (size_t)64 * DIC);
    cvt(mm_dt_w[i], w_dt_w[i], (size_t)DIC * DTRNK);
    cvt(mm_out_w[i], w_out_w[i], (size_t)DDIM * DIC);
  }
  cvt(c_w1, w_c_w1, (size_t)DI2 * DDIM);
  cvt(c_w2, w_c_w2, (size_t)DDIM * DIC);

  auto gemm = [&](const float* A, int lda, const unsigned short* Bw, const float* bias,
                  const float* src, float* C, int ldc, int M, int N, int K,
                  float alpha, int act) {
    dim3 g((N + 127) / 128, (M + 127) / 128);
    gemm_bf16_wmma<<<g, 256, 0, stream>>>(A, lda, Bw, bias, src, C, ldc, M, N, K, alpha, act);
  };
  auto ln = [&](const float* in, const float* g, const float* b, float* y) {
    ln_kernel<<<TTOK / 8, 256, 0, stream>>>(in, g, b, y);
  };
  const int EW = (TTOK * DIC + 255) / 256;

  // ================= FFM1: resid = x + 0.5*ffm(x) =================
  ln(x, ffm_g[0], ffm_b[0], lnbuf);
  gemm(lnbuf, DDIM, w_ffm_w1[0], ffm_b1[0], nullptr, buf2048, DI2, TTOK, DI2, DDIM, 1.f, 1);
  gemm(buf2048, DI2, w_ffm_w2[0], ffm_b2[0], x, resid, DDIM, TTOK, DDIM, DI2, 0.5f, 0);

  // ================= Mamba (bidirectional) =================
  ln(resid, m_ln_g, m_ln_b, lnbuf);
  for (int dir = 0; dir < 2; ++dir) {
    gemm(lnbuf, DDIM, w_in_w[dir], nullptr, nullptr, buf2048, DI2, TTOK, DI2, DDIM, 1.f, 0);
    mamba_conv_silu<<<EW, 256, 0, stream>>>(buf2048, mm_conv_w[dir], mm_conv_b[dir], b1024a, dir);
    gemm(b1024a, DIC, w_xproj[dir], nullptr, nullptr, bufS, 64, TTOK, 64, DIC, 1.f, 0);
    gemm(bufS, 64, w_dt_w[dir], mm_dt_b[dir], nullptr, b1024b, DIC, TTOK, DIC, DTRNK, 1.f, 2);
    mamba_scan_kernel<<<(BDIM * DIC + 255) / 256, 256, 0, stream>>>(
        b1024b, bufS, b1024a, buf2048, mm_Alog[dir], mm_D[dir], b1024c, dir);
    gemm(b1024c, DIC, w_out_w[dir], nullptr, resid, resid, DDIM, TTOK, DDIM, DIC, 1.f, 0);
  }

  // ================= CCM: resid += ccm(resid) =================
  ln(resid, c_g, c_b, lnbuf);
  gemm(lnbuf, DDIM, w_c_w1, c_b1, nullptr, buf2048, DI2, TTOK, DI2, DDIM, 1.f, 0);
  glu_kernel<<<EW, 256, 0, stream>>>(buf2048, b1024a);
  ccm_conv_bn_silu<<<EW, 256, 0, stream>>>(b1024a, c_dw, c_dwb, c_bng, c_bnb, c_bnm, c_bnv, b1024b);
  gemm(b1024b, DIC, w_c_w2, c_b2, resid, resid, DDIM, TTOK, DDIM, DIC, 1.f, 0);

  // ================= FFM2: resid += 0.5*ffm(resid) =================
  ln(resid, ffm_g[1], ffm_b[1], lnbuf);
  gemm(lnbuf, DDIM, w_ffm_w1[1], ffm_b1[1], nullptr, buf2048, DI2, TTOK, DI2, DDIM, 1.f, 1);
  gemm(buf2048, DI2, w_ffm_w2[1], ffm_b2[1], resid, resid, DDIM, TTOK, DDIM, DI2, 0.5f, 0);

  // ================= final LayerNorm -> out =================
  ln(resid, post_g, post_b, out);
}
